// AGITransformerLayer_19026705122059
// MI455X (gfx1250) — matrix operand order repaired and checked by reading, not compiled, not run on hardware
//
#include <hip/hip_runtime.h>
#include <math.h>

typedef __attribute__((ext_vector_type(16))) _Float16     v16h;
typedef __attribute__((ext_vector_type(8)))  float        v8f;
typedef __attribute__((ext_vector_type(4)))  unsigned int v4u;
typedef __attribute__((ext_vector_type(8)))  int          v8i;
typedef __attribute__((ext_vector_type(4)))  int          v4i;

#define WMMA_F16(a, b, c) \
  __builtin_amdgcn_wmma_f32_16x16x32_f16(false, (a), false, (b), (short)0, (c), false, false)

// CDNA5 ISA 7.12.2: 16-bit A-matrix 16x32 (MxK). Lane L holds row M=L&15.
//   i in [0,8):  K = (L<16 ? 0 : 8)  + i
//   i in [8,16): K = (L<16 ? 16 : 24) + (i-8)
__device__ __forceinline__ int a_k_of(int lane, int i) {
  return (i < 8) ? (((lane < 16) ? 0 : 8) + i)
                 : (((lane < 16) ? 16 : 24) + (i - 8));
}

// ---------------------------------------------------------------------------
// Tensor Data Mover: DMA a 2D f32 tile [tile_d1 rows x tile_d0 cols] from a
// row-major tensor (row stride = stride0 elements) into LDS at ldst (packed
// row-major). D# per CDNA5 ISA section 8 (groups 0/1, 2D). Per-wave op
// (EXEC ignored); completion tracked by TENSORcnt, in order per wave.
// ---------------------------------------------------------------------------
#if __has_builtin(__builtin_amdgcn_tensor_load_to_lds)
#define HAVE_TDM 1
__device__ __forceinline__ void tdm_load_2d_f32(const float* gsrc, void* ldst,
                                                unsigned tensor_d0, unsigned tensor_d1,
                                                unsigned tile_d0, unsigned tile_d1,
                                                unsigned stride0) {
  const unsigned long long ga = (unsigned long long)(uintptr_t)gsrc;
  const unsigned lds = (unsigned)(uintptr_t)ldst;  // generic->LDS offset (addr[31:0])
  v4u g0;
  g0[0] = 1u;                                          // count=1, user descriptor
  g0[1] = lds;                                         // lds_addr  (bits 63:32)
  g0[2] = (unsigned)(ga & 0xFFFFFFFFu);                // global_addr[31:0]
  g0[3] = (unsigned)((ga >> 32) & 0x01FFFFFFu) | (2u << 30);  // ga[56:32] | type=2
  v8i g1;
  g1[0] = (int)(2u << 16);                             // data_size=4B, mask=0, no flags
  g1[1] = (int)((tensor_d0 & 0xFFFFu) << 16);          // tensor_dim0[15:0] @ bits 63:48
  g1[2] = (int)((tensor_d0 >> 16) | ((tensor_d1 & 0xFFFFu) << 16));
  g1[3] = (int)((tensor_d1 >> 16) | (tile_d0 << 16));  // tile_dim0 @ bits 127:112
  g1[4] = (int)(tile_d1 & 0xFFFFu);                    // tile_dim1, tile_dim2=0
  g1[5] = (int)stride0;                                // tensor_dim0_stride[31:0]
  g1[6] = 0;
  g1[7] = 0;
  const v4i z4 = (v4i){0, 0, 0, 0};
#if defined(__clang_major__) && (__clang_major__ >= 23)
  const v8i z8 = (v8i){0, 0, 0, 0, 0, 0, 0, 0};
  __builtin_amdgcn_tensor_load_to_lds(g0, g1, z4, z4, z8, 0);
#else
  __builtin_amdgcn_tensor_load_to_lds(g0, g1, z4, z4, 0);
#endif
}
#else
#define HAVE_TDM 0
#endif

// ---------------------------------------------------------------------------
// Block-tiled GEMM: C[M,N] = A[M,K]*W[K,N] + bias.
// Block = 256 threads (8 waves) -> 128x64 tile. TDM double-buffers the
// A(128x32) / W(32x64) f32 stages in LDS: stage k+1 is in flight
// (s_wait_tensorcnt 2) while stage k feeds ds_load_b128 + cvt + WMMA.
// ---------------------------------------------------------------------------
__global__ void __launch_bounds__(256)
gemm_wmma_tiled(const float* __restrict__ A, const float* __restrict__ W,
                const float* __restrict__ bias, float* __restrict__ C,
                int M, int N, int Kdim) {
  __shared__ float Af[2][128][32];  // 2 x 16 KB
  __shared__ float Wf[2][32][64];   // 2 x  8 KB

  const int lane = threadIdx.x & 31;
  const int w    = threadIdx.x >> 5;
  const int m15  = lane & 15;
  const int hhf  = lane >> 4;
  const int mbase = blockIdx.x * 128;
  const int nbase = blockIdx.y * 64;
  const int nk = Kdim / 32;

  v8f acc[4];
#pragma unroll
  for (int t = 0; t < 4; ++t) acc[t] = (v8f){};

#if HAVE_TDM
  if (w == 0) {  // prologue: stage 0
    tdm_load_2d_f32(A + (size_t)mbase * Kdim, &Af[0][0][0],
                    (unsigned)Kdim, (unsigned)M, 32u, 128u, (unsigned)Kdim);
    tdm_load_2d_f32(W + nbase, &Wf[0][0][0],
                    (unsigned)N, (unsigned)Kdim, 64u, 32u, (unsigned)N);
  }
#endif

  for (int kk = 0; kk < nk; ++kk) {
    const int cur = kk & 1;
#if HAVE_TDM
    if (w == 0) {
      if (kk + 1 < nk) {  // prefetch stage k+1 into the other buffer
        const int k0n = (kk + 1) * 32;
        tdm_load_2d_f32(A + (size_t)mbase * Kdim + k0n, &Af[cur ^ 1][0][0],
                        (unsigned)Kdim, (unsigned)M, 32u, 128u, (unsigned)Kdim);
        tdm_load_2d_f32(W + (size_t)k0n * N + nbase, &Wf[cur ^ 1][0][0],
                        (unsigned)N, (unsigned)Kdim, 64u, 32u, (unsigned)N);
        __builtin_amdgcn_s_wait_tensorcnt(2);  // stage k done; k+1 in flight
      } else {
        __builtin_amdgcn_s_wait_tensorcnt(0);
      }
    }
#else
    {
      const int k0 = kk * 32;
      for (int idx = threadIdx.x; idx < 128 * 32 / 4; idx += 256) {
        const int r = idx >> 3, c = (idx & 7) << 2;
        *(float4*)&Af[cur][r][c] = *(const float4*)&A[(size_t)(mbase + r) * Kdim + k0 + c];
      }
      for (int idx = threadIdx.x; idx < 32 * 64 / 4; idx += 256) {
        const int r = idx >> 4, c = (idx & 15) << 2;
        *(float4*)&Wf[cur][r][c] = *(const float4*)&W[(size_t)(k0 + r) * N + nbase + c];
      }
    }
#endif
    __syncthreads();

    const int kb0 = (lane < 16) ? 0 : 8;
    const float* ar = &Af[cur][w * 16 + m15][0];
    v16h a;
#pragma unroll
    for (int i = 0; i < 8; ++i) a[i] = (_Float16)ar[kb0 + i];
#pragma unroll
    for (int i = 0; i < 8; ++i) a[8 + i] = (_Float16)ar[kb0 + 16 + i];

#pragma unroll
    for (int t = 0; t < 4; ++t) {
      const float* wr = &Wf[cur][0][t * 16 + m15];  // B frag: n = lane%16, k = a_k_of(lane, element)
      v16h b;
#pragma unroll
      for (int i = 0; i < 16; ++i) b[i] = (_Float16)wr[a_k_of(lane, i) * 64];
      acc[t] = WMMA_F16(a, b, acc[t]);
    }
    __syncthreads();  // consumers done before this buffer is re-filled
  }

#pragma unroll
  for (int t = 0; t < 4; ++t) {
    const int n = nbase + t * 16 + m15;
    const float bv = bias ? bias[n] : 0.0f;
#pragma unroll
    for (int r = 0; r < 8; ++r) {
      const int mm = mbase + w * 16 + r + hhf * 8;
      C[(size_t)mm * N + n] = acc[t][r] + bv;
    }
  }
}

// ---------------------------------------------------------------------------
// Narrow GEMM (N==16) for the gate projection: one wave per 16-row tile.
// ---------------------------------------------------------------------------
__global__ void gemm_wmma_n16(const float* __restrict__ A, const float* __restrict__ W,
                              const float* __restrict__ bias, float* __restrict__ C,
                              int M, int N, int Kdim) {
  const int lane = threadIdx.x & 31;
  const int wid  = blockIdx.x * (blockDim.x >> 5) + (threadIdx.x >> 5);
  if (wid * 16 >= M) return;
  const int m15 = lane & 15, hhf = lane >> 4;
  const float* Arow = A + (size_t)(wid * 16 + m15) * Kdim;
  v8f acc = (v8f){};
  for (int k0 = 0; k0 < Kdim; k0 += 32) {
    v16h a;
#pragma unroll
    for (int i = 0; i < 16; ++i) a[i] = (_Float16)Arow[k0 + a_k_of(lane, i)];
    const float* Wrow = W + (size_t)k0 * N + m15;
    v16h b;
#pragma unroll
    for (int i = 0; i < 16; ++i) b[i] = (_Float16)Wrow[(size_t)a_k_of(lane, i) * N];
    acc = WMMA_F16(a, b, acc);
  }
  const float bv = bias ? bias[m15] : 0.0f;
#pragma unroll
  for (int r = 0; r < 8; ++r)
    C[(size_t)(wid * 16 + r + hhf * 8) * N + m15] = acc[r] + bv;
}

// ---------------------------------------------------------------------------
// Flash attention. Block = 8 waves sharing one (batch, head); wave w owns the
// 16-query block qb0+w. K/V 32-key tiles are TDM double-buffered in LDS and
// shared by all 8 waves. Q fragments + sigmoid gate (with 1/sqrt(hd) folded)
// are hoisted to registers. QK^T and PV via WMMA; online softmax via
// shfl_xor reductions in 16-lane halves; P goes C-frag -> LDS -> A-frag
// fenced with s_wait_dscnt.
// ---------------------------------------------------------------------------
template <int HD>
__global__ void __launch_bounds__(256)
attn_wmma_kernel(const float* __restrict__ Q, const float* __restrict__ Kmat,
                 const float* __restrict__ V, const float* __restrict__ mod,
                 float* __restrict__ O, int S, int nh, int ld, int ldo) {
  constexpr int KC = HD / 32;
  constexpr int DC = HD / 16;
  __shared__ float Kf[2][32][HD];
  __shared__ float Vf[2][32][HD];
  __shared__ _Float16 psh[8][16][32];

  const int lane = threadIdx.x & 31;
  const int w    = threadIdx.x >> 5;
  const int qblocks = S / 16;
  const int gw0 = blockIdx.x * 8;           // 8 waves share (b,h); qblocks%8==0
  const int b   = gw0 / (nh * qblocks);
  const int h   = (gw0 / qblocks) % nh;
  const int qb  = (gw0 % qblocks) + w;
  const int m15 = lane & 15, hhf = lane >> 4;
  const float scale = rsqrtf((float)HD);

  // loop-invariant Q fragments (registers)
  const float* Qrow = Q + (size_t)(b * S + qb * 16 + m15) * ld + h * HD;
  v16h aq[KC];
#pragma unroll
  for (int kc = 0; kc < KC; ++kc)
#pragma unroll
    for (int i = 0; i < 16; ++i)
      aq[kc][i] = (_Float16)Qrow[kc * 32 + a_k_of(lane, i)];

  // loop-invariant per-row gate = scale * sigmoid-mod
  float gate[8];
#pragma unroll
  for (int r = 0; r < 8; ++r)
    gate[r] = scale * (mod ? mod[(size_t)(b * S + qb * 16 + r + hhf * 8) * nh + h] : 1.0f);

  float mrun[8], lrun[8];
  v8f accO[DC];
#pragma unroll
  for (int r = 0; r < 8; ++r) { mrun[r] = -1e30f; lrun[r] = 0.0f; }
#pragma unroll
  for (int j = 0; j < DC; ++j) accO[j] = (v8f){};

  const int nkb = S / 32;
#if HAVE_TDM
  if (w == 0) {  // prologue: stage 0 K/V tiles
    tdm_load_2d_f32(Kmat + (size_t)(b * S) * ld + h * HD, &Kf[0][0][0],
                    (unsigned)ld, (unsigned)S, (unsigned)HD, 32u, (unsigned)ld);
    tdm_load_2d_f32(V + (size_t)(b * S) * ld + h * HD, &Vf[0][0][0],
                    (unsigned)ld, (unsigned)S, (unsigned)HD, 32u, (unsigned)ld);
  }
#endif

  for (int kbi = 0; kbi < nkb; ++kbi) {
    const int cur = kbi & 1;
#if HAVE_TDM
    if (w == 0) {
      if (kbi + 1 < nkb) {
        const int kbn = (kbi + 1) * 32;
        tdm_load_2d_f32(Kmat + (size_t)(b * S + kbn) * ld + h * HD, &Kf[cur ^ 1][0][0],
                        (unsigned)ld, (unsigned)S, (unsigned)HD, 32u, (unsigned)ld);
        tdm_load_2d_f32(V + (size_t)(b * S + kbn) * ld + h * HD, &Vf[cur ^ 1][0][0],
                        (unsigned)ld, (unsigned)S, (unsigned)HD, 32u, (unsigned)ld);
        __builtin_amdgcn_s_wait_tensorcnt(2);
      } else {
        __builtin_amdgcn_s_wait_tensorcnt(0);
      }
    }
#else
    {
      const int kb = kbi * 32;
      for (int idx = threadIdx.x; idx < 32 * HD / 4; idx += 256) {
        const int r = idx / (HD / 4), c = (idx % (HD / 4)) * 4;
        *(float4*)&Kf[cur][r][c] =
            *(const float4*)&Kmat[(size_t)(b * S + kb + r) * ld + h * HD + c];
        *(float4*)&Vf[cur][r][c] =
            *(const float4*)&V[(size_t)(b * S + kb + r) * ld + h * HD + c];
      }
    }
#endif
    __syncthreads();

    // ---- S tile = Q[16,HD] x K^T[HD,32] from LDS ----
    v8f c0 = (v8f){}, c1 = (v8f){};
#pragma unroll
    for (int kc = 0; kc < KC; ++kc) {
      v16h bk0, bk1;
#pragma unroll
      for (int i = 0; i < 16; ++i) {
        bk0[i] = (_Float16)Kf[cur][m15][kc * 32 + a_k_of(lane, i)];       // n=lane%16(key), k=a_k_of(lane,i)(d)
        bk1[i] = (_Float16)Kf[cur][16 + m15][kc * 32 + a_k_of(lane, i)];
      }
      c0 = WMMA_F16(aq[kc], bk0, c0);
      c1 = WMMA_F16(aq[kc], bk1, c1);
    }

    // ---- gated online softmax; stash P tile into LDS ----
#pragma unroll
    for (int r = 0; r < 8; ++r) {
      float s0 = c0[r] * gate[r], s1 = c1[r] * gate[r];
      float mx = fmaxf(s0, s1);
#pragma unroll
      for (int d = 1; d < 16; d <<= 1) mx = fmaxf(mx, __shfl_xor(mx, d, 32));
      const float mnew = fmaxf(mrun[r], mx);
      const float corr = __expf(mrun[r] - mnew);
      mrun[r] = mnew;
      const float p0 = __expf(s0 - mnew), p1 = __expf(s1 - mnew);
      float rs = p0 + p1;
#pragma unroll
      for (int d = 1; d < 16; d <<= 1) rs += __shfl_xor(rs, d, 32);
      lrun[r] = lrun[r] * corr + rs;
#pragma unroll
      for (int j = 0; j < DC; ++j) accO[j][r] *= corr;

      const int mrow = r + hhf * 8;
      psh[w][mrow][m15]      = (_Float16)p0;
      psh[w][mrow][16 + m15] = (_Float16)p1;
    }
    asm volatile("s_wait_dscnt 0" ::: "memory");  // wave-local P transpose fence

    v16h pa;
#pragma unroll
    for (int i = 0; i < 16; ++i) pa[i] = psh[w][m15][a_k_of(lane, i)];

    // ---- O += P[16,32] x V[32,HD] from LDS ----
#pragma unroll
    for (int j = 0; j < DC; ++j) {
      v16h bv;
#pragma unroll
      for (int i = 0; i < 16; ++i) bv[i] = (_Float16)Vf[cur][a_k_of(lane, i)][j * 16 + m15];
      accO[j] = WMMA_F16(pa, bv, accO[j]);
    }
    __syncthreads();  // all waves done with this K/V buffer before refill
  }

#pragma unroll
  for (int r = 0; r < 8; ++r) {
    const float inv = 1.0f / lrun[r];
    const size_t orow = (size_t)(b * S + qb * 16 + r + hhf * 8) * ldo + h * HD;
#pragma unroll
    for (int j = 0; j < DC; ++j) O[orow + j * 16 + m15] = accO[j][r] * inv;
  }
}

__global__ void amvec_kernel(const float* __restrict__ cons,
                             const float* __restrict__ am_W,
                             const float* __restrict__ am_b,
                             float* __restrict__ amv) {
  const int n = threadIdx.x;  // 16 threads
  float s = am_b[n];
  for (int k = 0; k < 16; ++k) s += cons[k] * am_W[k * 16 + n];
  amv[n] = s;
}

__global__ void mod_kernel(float* __restrict__ logits,
                           const float* __restrict__ amv, int n) {
  const int i = blockIdx.x * blockDim.x + threadIdx.x;
  if (i < n) {
    const float x = logits[i] + amv[i & 15];
    logits[i] = 1.0f / (1.0f + __expf(-x));
  }
}

__global__ void blend_kernel(float* __restrict__ x, const float* __restrict__ y,
                             float wa, float wb, int n) {
  const int i = blockIdx.x * blockDim.x + threadIdx.x;
  if (i < n) x[i] = x[i] * wa + y[i] * wb;
}

extern "C" void kernel_launch(void* const* d_in, const int* in_sizes, int n_in,
                              void* d_out, int out_size, void* d_ws, size_t ws_size,
                              hipStream_t stream) {
  const float* X      = (const float*)d_in[0];
  const float* cons   = (const float*)d_in[1];
  const float* Wq     = (const float*)d_in[2];
  const float* bq     = (const float*)d_in[3];
  const float* Wk     = (const float*)d_in[4];
  const float* bk     = (const float*)d_in[5];
  const float* Wv     = (const float*)d_in[6];
  const float* bv     = (const float*)d_in[7];
  const float* cg_W   = (const float*)d_in[8];
  const float* cg_b   = (const float*)d_in[9];
  const float* am_W   = (const float*)d_in[10];
  const float* am_b   = (const float*)d_in[11];
  const float* caWin  = (const float*)d_in[12];
  const float* caBin  = (const float*)d_in[13];
  const float* caWout = (const float*)d_in[14];
  const float* caBout = (const float*)d_in[15];
  const float* mcWin  = (const float*)d_in[16];
  const float* mcBin  = (const float*)d_in[17];
  const float* mcWout = (const float*)d_in[18];
  const float* mcBout = (const float*)d_in[19];
  const float* Wo     = (const float*)d_in[20];
  const float* bo     = (const float*)d_in[21];
  float* out = (float*)d_out;

  const int Bsz = 4, S = 1024, H = 1024, NH = 16;
  const int M = Bsz * S;

  float* bufA   = (float*)d_ws;                 // M*3H : QKV / packed in_proj
  float* bufB   = bufA + (size_t)M * 3 * H;     // M*H  : running ctx
  float* bufC   = bufB + (size_t)M * H;         // M*H  : branch attention out
  float* bufD   = bufC + (size_t)M * H;         // M*H  : branch out_proj
  float* bufMod = bufD + (size_t)M * H;         // M*NH : gate
  float* amv    = bufMod + (size_t)M * NH;      // 16

  float* Qb = bufA;
  float* Kb = bufA + (size_t)M * H;
  float* Vb = bufA + (size_t)2 * M * H;

  const dim3 blk(256);
  auto gemm = [&](const float* A, const float* W, const float* bias, float* C,
                  int N, int Kd) {
    dim3 grid(M / 128, N / 64);
    gemm_wmma_tiled<<<grid, blk, 0, stream>>>(A, W, bias, C, M, N, Kd);
  };

  // 1) main-branch Q/K/V projections
  gemm(X, Wq, bq, Qb, H, H);
  gemm(X, Wk, bk, Kb, H, H);
  gemm(X, Wv, bv, Vb, H, H);

  // 2) consciousness gate: sigmoid(X@cg_W + cg_b + cons@am_W + am_b)
  gemm_wmma_n16<<<(M / 16) / 8, blk, 0, stream>>>(X, cg_W, cg_b, bufMod, M, NH, H);
  amvec_kernel<<<1, 16, 0, stream>>>(cons, am_W, am_b, amv);
  mod_kernel<<<(M * NH + 255) / 256, blk, 0, stream>>>(bufMod, amv, M * NH);

  // 3) modulated main attention -> bufB
  {
    const int waves = Bsz * NH * (S / 16);
    attn_wmma_kernel<64><<<waves / 8, blk, 0, stream>>>(Qb, Kb, Vb, bufMod, bufB,
                                                        S, NH, H, H);
  }

  // 4) causal MHA branch (4 heads, hd=256), blend 0.3/0.7
  gemm(X, caWin, caBin, bufA, 3 * H, H);
  {
    const int waves = Bsz * 4 * (S / 16);
    attn_wmma_kernel<256><<<waves / 8, blk, 0, stream>>>(bufA, bufA + H, bufA + 2 * H,
                                                         nullptr, bufC, S, 4, 3 * H, H);
  }
  gemm(bufC, caWout, caBout, bufD, H, H);
  blend_kernel<<<(M * H + 255) / 256, blk, 0, stream>>>(bufB, bufD, 0.3f, 0.7f, M * H);

  // 5) metacognitive MHA branch (4 heads, hd=256), blend 0.85/0.15
  gemm(bufB, mcWin, mcBin, bufA, 3 * H, H);
  {
    const int waves = Bsz * 4 * (S / 16);
    attn_wmma_kernel<256><<<waves / 8, blk, 0, stream>>>(bufA, bufA + H, bufA + 2 * H,
                                                         nullptr, bufC, S, 4, 3 * H, H);
  }
  gemm(bufC, mcWout, mcBout, bufD, H, H);
  blend_kernel<<<(M * H + 255) / 256, blk, 0, stream>>>(bufB, bufD, 0.85f, 0.15f, M * H);

  // 6) output projection -> d_out
  gemm(bufB, Wo, bo, out, H, H);
}
